// MultiHeadAttention_79671643340806
// MI455X (gfx1250) — matrix operand order repaired and checked by reading, not compiled, not run on hardware
//
#include <hip/hip_runtime.h>
#include <hip/hip_bf16.h>

#define S_LEN   2048
#define D_MODEL 1024
#define NHEADS  16
#define HDEPTH  64
#define BATCH   2
#define MTOT    (BATCH * S_LEN)   // 4096 total rows

typedef __attribute__((ext_vector_type(16))) __bf16 v16bf;
typedef __attribute__((ext_vector_type(8)))  float  v8f;

union BfFrag { unsigned u[8]; v16bf v; };
union FAcc   { float f[8];    v8f   v; };

__device__ __forceinline__ unsigned short f32_to_bf16(float x) {
  unsigned u = __builtin_bit_cast(unsigned, x);
  unsigned r = u + 0x7FFFu + ((u >> 16) & 1u);   // round-to-nearest-even
  return (unsigned short)(r >> 16);
}
__device__ __forceinline__ unsigned pack_bf16x2(float lo, float hi) {
  return (unsigned)f32_to_bf16(lo) | ((unsigned)f32_to_bf16(hi) << 16);
}

// issue one wave-wide async global->LDS 16B/lane copy (ASYNCcnt++)
__device__ __forceinline__ void async_copy_b128(unsigned lds_addr, const void* gptr) {
  asm volatile("global_load_async_to_lds_b128 %0, %1, off"
               :: "v"(lds_addr), "v"((unsigned long long)gptr) : "memory");
}

// ---------------------------------------------------------------------------
// Y[4096x1024] = X[4096x1024] @ W[1024x1024] + bias, bf16 WMMA, f32 accum.
// X tile staged via async global->LDS DMA, double buffered (ASYNCcnt pipeline);
// W tile staged via VGPRs (needs f32->bf16 transpose), loads issued one step
// ahead so the loadcnt wait lands behind the WMMAs.
// OUT_MODE 0: bf16 head-major [B*H][S][64]          (Q/K projections)
// OUT_MODE 1: bf16 head-major transposed [B*H][64][S] (V projection)
// OUT_MODE 2: f32 row-major [4096][1024]            (final output projection)
// ---------------------------------------------------------------------------
template<bool X_BF16, int OUT_MODE>
__global__ __launch_bounds__(256) void proj_gemm(const void* __restrict__ Xv,
    const float* __restrict__ W, const float* __restrict__ bias,
    void* __restrict__ Out)
{
  // X staging: f32 [2][128][36] (pad 4 -> 144B rows, 16B aligned)
  //            bf16 [2][128][40] (pad 8 -> 80B rows, 16B aligned)
  __shared__ __align__(16) unsigned char smemX[X_BF16 ? (2*128*40*2) : (2*128*36*4)];
  __shared__ unsigned short Wst[2][64][34];   // W tile transposed [n][k], bf16

  const int tid  = threadIdx.x;
  const int lane = tid & 31, lr = lane & 15, lh = lane >> 4;
  const int wave = tid >> 5;
  const int wm = wave & 3, wn = wave >> 2;            // 4x2 wave grid
  const int m0 = blockIdx.y * 128, n0 = blockIdx.x * 64;
  const float*          Xf = (const float*)Xv;
  const unsigned short* Xb = (const unsigned short*)Xv;
  const unsigned xlds0 = (unsigned)(unsigned long long)(void*)&smemX[0];
  const int XBUFB = X_BF16 ? (128*40*2) : (128*36*4);
  const int XROW  = X_BF16 ? 40 : 36;                  // row stride in elements
  // async instructions per wave per tile (for s_wait_asynccnt thresholds)
  //   f32: 16KB / (8 waves * 512B) = 4 ; bf16: 8KB -> 2

  FAcc c[2][2];
  for (int i = 0; i < 2; ++i)
    for (int j = 0; j < 2; ++j)
      for (int r = 0; r < 8; ++r) c[i][j].f[r] = 0.f;

  // -- async-issue the X tile for k-step k0 into buffer `buf`
  auto issueX = [&](int k0, int buf) {
    if (X_BF16) {
      #pragma unroll
      for (int it = 0; it < 2; ++it) {             // 8 ushort = 16B per lane
        int e = (it * 256 + tid) * 8;
        int r = e >> 5, cc = e & 31;
        async_copy_b128(xlds0 + buf * XBUFB + (r * XROW + cc) * 2,
                        Xb + (size_t)(m0 + r) * D_MODEL + k0 + cc);
      }
    } else {
      #pragma unroll
      for (int it = 0; it < 4; ++it) {             // 4 f32 = 16B per lane
        int f = (it * 256 + tid) * 4;
        int r = f >> 5, cc = f & 31;
        async_copy_b128(xlds0 + buf * XBUFB + (r * XROW + cc) * 4,
                        Xf + (size_t)(m0 + r) * D_MODEL + k0 + cc);
      }
    }
  };

  float wreg[8];                                   // W tile in flight (8/thread)
  auto loadW = [&](int k0) {
    #pragma unroll
    for (int t = 0; t < 8; ++t) {
      int e = t * 256 + tid;
      int kr = e >> 6, nc = e & 63;
      wreg[t] = W[(size_t)(k0 + kr) * D_MODEL + n0 + nc];
    }
  };
  auto writeW = [&](int buf) {
    #pragma unroll
    for (int t = 0; t < 8; ++t) {
      int e = t * 256 + tid;
      int kr = e >> 6, nc = e & 63;
      Wst[buf][nc][kr] = f32_to_bf16(wreg[t]);
    }
  };

  loadW(0);
  issueX(0, 0);

  for (int i = 0; i < D_MODEL / 32; ++i) {
    const int k0  = i * 32;
    const int buf = i & 1;
    const bool hasNext = (k0 + 32 < D_MODEL);

    writeW(buf);                                   // W tile i -> LDS (waits its loads)
    if (hasNext) {
      issueX(k0 + 32, buf ^ 1);                    // DMA next X tile (other buffer)
      loadW(k0 + 32);                              // next W tile into VGPRs (in flight)
      if (k0 + 64 < D_MODEL)
        __builtin_prefetch(&W[(size_t)(k0 + 64) * D_MODEL + n0 + (tid & 63)], 0, 0);
      // tile i complete when only the just-issued tile i+1 remains outstanding
      asm volatile("s_wait_asynccnt %0" :: "n"(X_BF16 ? 2 : 4) : "memory");
    } else {
      asm volatile("s_wait_asynccnt 0" ::: "memory");
    }
    __syncthreads();                               // tile i (X async + W ds) visible

    // A fragments: 16x32 bf16, lane row = M, VGPR v -> K=(v/4)*16 + lh*8 + (v%4)*2
    BfFrag a[2], b[2];
    #pragma unroll
    for (int ii = 0; ii < 2; ++ii) {
      int row = wm * 32 + ii * 16 + lr;
      if (X_BF16) {
        const unsigned short* xrow =
            (const unsigned short*)smemX + buf * (128 * 40) + row * XROW;
        #pragma unroll
        for (int v = 0; v < 8; ++v) {
          int kk = (v >> 2) * 16 + lh * 8 + (v & 3) * 2;
          a[ii].u[v] = *(const unsigned*)&xrow[kk];
        }
      } else {
        const float* xrow = (const float*)smemX + buf * (128 * 36) + row * XROW;
        #pragma unroll
        for (int v = 0; v < 8; ++v) {
          int kk = (v >> 2) * 16 + lh * 8 + (v & 3) * 2;
          a[ii].u[v] = pack_bf16x2(xrow[kk], xrow[kk + 1]);
        }
      }
    }
    // B fragments: 32x16 bf16, lane col = N, VGPR v -> K=(v/4)*16 + lh*8 + (v%4)*2
    #pragma unroll
    for (int j = 0; j < 2; ++j) {
      int n = wn * 32 + j * 16 + lr;
      #pragma unroll
      for (int v = 0; v < 8; ++v) {
        int kk = (v >> 2) * 16 + lh * 8 + (v & 3) * 2;
        b[j].u[v] = *(const unsigned*)&Wst[buf][n][kk];
      }
    }
    #pragma unroll
    for (int ii = 0; ii < 2; ++ii)
      #pragma unroll
      for (int j = 0; j < 2; ++j)
        c[ii][j].v = __builtin_amdgcn_wmma_f32_16x16x32_bf16(
            false, a[ii].v, false, b[j].v, (short)0, c[ii][j].v, false, false);
    __syncthreads();                               // everyone done reading buf
  }

  // epilogue: C layout -> M = r + 8*lh, N = lr
  for (int i = 0; i < 2; ++i)
    for (int j = 0; j < 2; ++j) {
      int gcol = n0 + wn * 32 + j * 16 + lr;
      float bv = bias[gcol];
      #pragma unroll
      for (int r = 0; r < 8; ++r) {
        int grow = m0 + wm * 32 + i * 16 + r + 8 * lh;
        float val = c[i][j].f[r] + bv;
        if (OUT_MODE == 2) {
          ((float*)Out)[(size_t)grow * D_MODEL + gcol] = val;
        } else {
          int bb = grow >> 11, srow = grow & (S_LEN - 1);
          int h = gcol >> 6,  dd = gcol & (HDEPTH - 1);
          unsigned short ov = f32_to_bf16(val);
          if (OUT_MODE == 0)
            ((unsigned short*)Out)[((size_t)(bb * NHEADS + h) * S_LEN + srow) * HDEPTH + dd] = ov;
          else
            ((unsigned short*)Out)[((size_t)(bb * NHEADS + h) * HDEPTH + dd) * S_LEN + srow] = ov;
        }
      }
    }
}

// ---------------------------------------------------------------------------
// Flash attention, causal. One wave owns 16 q rows; 4 waves/block share nothing
// in the hot loop (no block barriers). logits^T = K*Q^T so the softmax axis is
// per-lane registers + one shfl_xor(16); out^T += V^T * P^T via WMMA.
// ---------------------------------------------------------------------------
__global__ __launch_bounds__(128) void flash_attn(
    const unsigned short* __restrict__ qp,  // [B*H][S][64] bf16
    const unsigned short* __restrict__ kp,  // [B*H][S][64] bf16
    const unsigned short* __restrict__ vt,  // [B*H][64][S] bf16 (transposed)
    unsigned short* __restrict__ ao)        // [B*S][1024] bf16 (heads merged)
{
  __shared__ unsigned short Pst[4][16][34]; // wave-private P tile [q][k]

  const int lane = threadIdx.x & 31, lr = lane & 15, lh = lane >> 4;
  const int wave = threadIdx.x >> 5;
  const int bh = blockIdx.y;
  const int qbase = blockIdx.x * 64 + wave * 16;
  const size_t qkBase = (size_t)bh * S_LEN * HDEPTH;
  const size_t vBase  = (size_t)bh * HDEPTH * S_LEN;

  // Q fragments (B operand of K*Q^T): lane col = q row, VGPR v -> d pair
  BfFrag qf[2];
  {
    const unsigned* qrow = (const unsigned*)(qp + qkBase + (size_t)(qbase + lr) * HDEPTH);
    #pragma unroll
    for (int dh = 0; dh < 2; ++dh)
      #pragma unroll
      for (int v = 0; v < 8; ++v) {
        int kk = (v >> 2) * 16 + lh * 8 + (v & 3) * 2;
        qf[dh].u[v] = qrow[(dh * 32 + kk) >> 1];
      }
  }

  FAcc o[4];
  for (int dt = 0; dt < 4; ++dt)
    for (int r = 0; r < 8; ++r) o[dt].f[r] = 0.f;
  float run_max = -1e30f, run_sum = 0.f;
  const int qg = qbase + lr;          // this lane's q index (column of logits^T)
  const float scale = 0.03125f;       // 1/sqrt(D_MODEL) = 1/32

  for (int j = 0; j <= qbase + 15; j += 32) {   // causal: only k-tiles <= q
    // --- logits^T: two 16x16 tiles (k rows j..j+15, j+16..j+31) ---
    FAcc ct[2];
    for (int t = 0; t < 2; ++t)
      for (int r = 0; r < 8; ++r) ct[t].f[r] = 0.f;
    #pragma unroll
    for (int t = 0; t < 2; ++t) {
      const unsigned* krow =
          (const unsigned*)(kp + qkBase + (size_t)(j + t * 16 + lr) * HDEPTH);
      #pragma unroll
      for (int dh = 0; dh < 2; ++dh) {
        BfFrag kf;
        #pragma unroll
        for (int v = 0; v < 8; ++v) {
          int kk = (v >> 2) * 16 + lh * 8 + (v & 3) * 2;
          kf.u[v] = krow[(dh * 32 + kk) >> 1];
        }
        ct[t].v = __builtin_amdgcn_wmma_f32_16x16x32_bf16(
            false, kf.v, false, qf[dh].v, (short)0, ct[t].v, false, false);
      }
    }
    // --- scale + causal mask + online softmax (all values in lane have col q) ---
    float p[16];
    float mt = -1e30f;
    #pragma unroll
    for (int t = 0; t < 2; ++t)
      #pragma unroll
      for (int r = 0; r < 8; ++r) {
        int kg = j + t * 16 + r + 8 * lh;
        float lg = ct[t].f[r] * scale + (kg > qg ? -1e9f : 0.f);
        p[t * 8 + r] = lg;
        mt = fmaxf(mt, lg);
      }
    mt = fmaxf(mt, __shfl_xor(mt, 16, 32));     // other half-wave: same q, k+8
    float nm = fmaxf(run_max, mt);
    float alpha = __expf(run_max - nm);
    run_max = nm;
    float st = 0.f;
    #pragma unroll
    for (int t = 0; t < 2; ++t)
      #pragma unroll
      for (int r = 0; r < 8; ++r) {
        float pe = __expf(p[t * 8 + r] - nm);
        st += pe;
        Pst[wave][lr][t * 16 + r + 8 * lh] = f32_to_bf16(pe);
      }
    st += __shfl_xor(st, 16, 32);
    run_sum = run_sum * alpha + st;
    #pragma unroll
    for (int dt = 0; dt < 4; ++dt)
      #pragma unroll
      for (int r = 0; r < 8; ++r) o[dt].f[r] *= alpha;

    // --- P^T B-fragment from wave-private LDS (same-wave DS ops are in order) ---
    BfFrag pf;
    #pragma unroll
    for (int v = 0; v < 8; ++v) {
      int kk = (v >> 2) * 16 + lh * 8 + (v & 3) * 2;
      pf.u[v] = *(const unsigned*)&Pst[wave][lr][kk];
    }
    // --- out^T += V^T * P^T : 4 d-tiles of 16 rows each ---
    #pragma unroll
    for (int dt = 0; dt < 4; ++dt) {
      const unsigned* vrow =
          (const unsigned*)(vt + vBase + (size_t)(dt * 16 + lr) * S_LEN + j);
      BfFrag vf;
      #pragma unroll
      for (int v = 0; v < 8; ++v) {
        int kk = (v >> 2) * 16 + lh * 8 + (v & 3) * 2;
        vf.u[v] = vrow[kk >> 1];
      }
      o[dt].v = __builtin_amdgcn_wmma_f32_16x16x32_bf16(
          false, vf.v, false, pf.v, (short)0, o[dt].v, false, false);
    }
  }

  // normalize and write merged-head bf16 rows (pairs packed to 4B stores)
  float inv = 1.f / run_sum;
  const int b = bh >> 4, h = bh & 15;
  unsigned short* orow =
      ao + ((size_t)(b * S_LEN) + qbase + lr) * D_MODEL + h * HDEPTH;
  #pragma unroll
  for (int dt = 0; dt < 4; ++dt)
    #pragma unroll
    for (int r = 0; r < 8; r += 2) {
      int dd = dt * 16 + r + 8 * lh;
      *(unsigned*)&orow[dd] = pack_bf16x2(o[dt].f[r] * inv, o[dt].f[r + 1] * inv);
    }
}

// ---------------------------------------------------------------------------
extern "C" void kernel_launch(void* const* d_in, const int* in_sizes, int n_in,
                              void* d_out, int out_size, void* d_ws, size_t ws_size,
                              hipStream_t stream) {
  (void)in_sizes; (void)n_in; (void)out_size; (void)ws_size;
  const float* K  = (const float*)d_in[0];
  const float* V  = (const float*)d_in[1];
  const float* Q  = (const float*)d_in[2];
  /* d_in[3] = mask: strict-upper-triangular causal — applied analytically */
  const float* Wk = (const float*)d_in[4];
  const float* bk = (const float*)d_in[5];
  const float* Wv = (const float*)d_in[6];
  const float* bv = (const float*)d_in[7];
  const float* Wq = (const float*)d_in[8];
  const float* bq = (const float*)d_in[9];
  const float* Wo = (const float*)d_in[10];
  const float* bo = (const float*)d_in[11];

  const size_t SZ = (size_t)MTOT * D_MODEL * sizeof(unsigned short); // 8 MiB each
  unsigned short* qp  = (unsigned short*)d_ws;                       // q proj bf16
  unsigned short* kp  = (unsigned short*)((char*)d_ws + SZ);         // k proj bf16
  unsigned short* vtp = (unsigned short*)((char*)d_ws + 2 * SZ);     // v proj^T bf16
  unsigned short* ao  = (unsigned short*)((char*)d_ws + 3 * SZ);     // attn out bf16

  dim3 ggrid(D_MODEL / 64, MTOT / 128);   // (N tiles, M tiles)
  proj_gemm<false, 0><<<ggrid, 256, 0, stream>>>(Q, Wq, bq, qp);
  proj_gemm<false, 0><<<ggrid, 256, 0, stream>>>(K, Wk, bk, kp);
  proj_gemm<false, 1><<<ggrid, 256, 0, stream>>>(V, Wv, bv, vtp);
  flash_attn<<<dim3(S_LEN / 64, BATCH * NHEADS), 128, 0, stream>>>(qp, kp, vtp, ao);
  proj_gemm<true, 2><<<ggrid, 256, 0, stream>>>(ao, Wo, bo, d_out);
}
